// SingleLayerGRU_71468255806046
// MI455X (gfx1250) — matrix operand order repaired and checked by reading, not compiled, not run on hardware
//
#include <hip/hip_runtime.h>
#include <hip/hip_bf16.h>
#include <stdint.h>

// ---------------- problem constants ----------------
#define BATCH 64
#define SEQ   512
#define FIN   256
#define HID   512
#define KTOT  (FIN + HID)   // 768 = concat K
#define NFZ   (2*HID)       // 1024
#define FOUT  128
#define NWG   16            // persistent workgroups

// LDS staging: per-WG weight slices, resident for the whole time loop.
//   sWfz: 64 cols x 768 K  (96 KB)   at elem offset 0
//   sWr : 32 cols x 768 K  (48 KB)   at elem offset 64*KTOT
#define LDS_WFZ_ELEMS (64 * KTOT)
#define LDS_WR_ELEMS  (32 * KTOT)
#define LDS_BYTES     ((LDS_WFZ_ELEMS + LDS_WR_ELEMS) * 2)   // 147456

typedef __attribute__((ext_vector_type(16))) __bf16 v16bf;
typedef __attribute__((ext_vector_type(8)))  float  v8f;

union Frag128 { v16bf v; uint4 q[2]; };

// A fragment (16x32 bf16): per lane 32 contiguous bytes at byte
// offset 32*half within the row — explicit b128 loads.
__device__ __forceinline__ v16bf load_fragA(const __bf16* __restrict__ base,
                                            int row, int ld, int kbase, int half) {
  Frag128 f;
  const unsigned char* p =
      (const unsigned char*)base + ((size_t)row * ld + kbase) * 2 + 32 * half;
  f.q[0] = *(const uint4*)(p);
  f.q[1] = *(const uint4*)(p + 16);
  return f.v;
}

// B fragment (32x16 bf16) from a TRANSPOSED matrix (row = output column N):
// per lane 32 contiguous bytes at byte offset 32*half within the row.
__device__ __forceinline__ v16bf load_fragB(const __bf16* __restrict__ base,
                                            int row, int ld, int kbase, int half) {
  Frag128 f;
  const unsigned char* p =
      (const unsigned char*)base + ((size_t)row * ld + kbase) * 2 + 32 * half;
  f.q[0] = *(const uint4*)(p);
  f.q[1] = *(const uint4*)(p + 16);
  return f.v;
}

// B fragment from the LDS weight stash (elemBase = local_row*KTOT + kbase).
// All bases are 16B aligned by construction -> 2x ds_load_b128.
__device__ __forceinline__ v16bf lds_fragB(int elemBase, int half) {
  extern __shared__ __attribute__((aligned(16))) __bf16 lds_w[];
  Frag128 f;
  const uint4* p = (const uint4*)(lds_w + elemBase + 16 * half);
  f.q[0] = p[0];
  f.q[1] = p[1];
  return f.v;
}

__device__ __forceinline__ v8f wmma_bf16(v16bf a, v16bf b, v8f c) {
  return __builtin_amdgcn_wmma_f32_16x16x32_bf16(
      false, a, false, b, (short)0, c, false, false);
}

// ---------------- device-wide barrier for NWG resident blocks ----------------
__device__ __forceinline__ void grid_barrier(int* bar) {
  __syncthreads();
  if (threadIdx.x == 0) {
    __threadfence();
    int gen = __hip_atomic_load(&bar[1], __ATOMIC_RELAXED, __HIP_MEMORY_SCOPE_AGENT);
    if (__hip_atomic_fetch_add(&bar[0], 1, __ATOMIC_ACQ_REL,
                               __HIP_MEMORY_SCOPE_AGENT) == NWG - 1) {
      __hip_atomic_store(&bar[0], 0, __ATOMIC_RELAXED, __HIP_MEMORY_SCOPE_AGENT);
      __hip_atomic_fetch_add(&bar[1], 1, __ATOMIC_RELEASE, __HIP_MEMORY_SCOPE_AGENT);
    } else {
      while (__hip_atomic_load(&bar[1], __ATOMIC_ACQUIRE,
                               __HIP_MEMORY_SCOPE_AGENT) == gen)
        __builtin_amdgcn_s_sleep(1);
    }
    __threadfence();
  }
  __syncthreads();
}

// ---------------- prep kernels (fully parallel) ----------------
__global__ void gru_cvt_u(const float* __restrict__ u, __bf16* __restrict__ ub, int n) {
  int i = (blockIdx.x * blockDim.x + threadIdx.x) * 4;
  if (i >= n) return;
  float4 v = *(const float4*)(u + i);
  ub[i + 0] = (__bf16)v.x; ub[i + 1] = (__bf16)v.y;
  ub[i + 2] = (__bf16)v.z; ub[i + 3] = (__bf16)v.w;
}

__global__ void gru_prep_w(const float* __restrict__ kfz, const float* __restrict__ kr,
                           const float* __restrict__ wout,
                           __bf16* __restrict__ WfzT, __bf16* __restrict__ WrT,
                           __bf16* __restrict__ WoutT) {
  int i = blockIdx.x * blockDim.x + threadIdx.x;
  if (i < NFZ * KTOT) {
    int n = i / KTOT, k = i % KTOT;
    WfzT[i] = (__bf16)kfz[k * NFZ + n];
    return;
  }
  i -= NFZ * KTOT;
  if (i < HID * KTOT) {
    int n = i / KTOT, k = i % KTOT;
    WrT[i] = (__bf16)kr[k * HID + n];
    return;
  }
  i -= HID * KTOT;
  if (i < FOUT * HID) {
    int n = i / HID, k = i % HID;
    WoutT[i] = (__bf16)wout[k * FOUT + n];
  }
}

__global__ void gru_init_x(const float* __restrict__ x0, float* __restrict__ x_f32,
                           __bf16* __restrict__ x_bf, int* __restrict__ bar) {
  int i = blockIdx.x * blockDim.x + threadIdx.x;
  if (i < BATCH * HID) {
    float v = x0[i];
    x_f32[i] = v;
    x_bf[i]  = (__bf16)v;
  }
  if (i == 0) { bar[0] = 0; bar[1] = 0; }
}

// ---------------- persistent recurrent kernel (16 WGs, 8 wave32 each) ----------------
__global__ void __launch_bounds__(256, 1) gru_recurrent(
    const __bf16* __restrict__ u_bf,
    const __bf16* __restrict__ WfzT, const __bf16* __restrict__ WrT,
    const float* __restrict__ bias_fz, const float* __restrict__ bias_r,
    float*  __restrict__ x_f32,   // [2][BATCH*HID] ping-pong
    __bf16* __restrict__ x_bf,    // [2][BATCH*HID] ping-pong
    __bf16* __restrict__ fx_bf,   // [BATCH*HID]
    float*  __restrict__ z_f32,   // [BATCH*HID]
    __bf16* __restrict__ xhist,   // [SEQ][BATCH*HID]
    int* __restrict__ bar) {
  const int wg   = blockIdx.x;
  const int tid  = threadIdx.x;
  const int w    = tid >> 5;        // wave 0..7
  const int half = (tid >> 4) & 1;
  const int idx  = tid & 15;
  const int mt   = w & 3;           // batch M-tile 0..3
  const int mrow = mt * 16 + idx;

  // ---- stage this WG's weight slices into LDS (once, async DMA path) ----
  {
    const __bf16* srcFz = WfzT + (size_t)wg * 64 * KTOT;  // 64 contiguous rows
    const __bf16* srcR  = WrT  + (size_t)wg * 32 * KTOT;  // 32 contiguous rows
    for (int i = tid; i < LDS_WFZ_ELEMS / 8; i += 256) {  // 16B per lane per op
      unsigned lds_off = (unsigned)(i * 16);
      unsigned long long ga = (unsigned long long)(uintptr_t)(srcFz + i * 8);
      asm volatile("global_load_async_to_lds_b128 %0, %1, off"
                   :: "v"(lds_off), "v"(ga) : "memory");
    }
    for (int i = tid; i < LDS_WR_ELEMS / 8; i += 256) {
      unsigned lds_off = (unsigned)(LDS_WFZ_ELEMS * 2 + i * 16);
      unsigned long long ga = (unsigned long long)(uintptr_t)(srcR + i * 8);
      asm volatile("global_load_async_to_lds_b128 %0, %1, off"
                   :: "v"(lds_off), "v"(ga) : "memory");
    }
    asm volatile("s_wait_asynccnt 0x0" ::: "memory");
    __syncthreads();
  }

  const int ntl  = (w >> 2) * 2;            // phase A: 2 N-tiles per wave
  const int lnA0 = ((ntl + 0) * 16 + idx);  // local B rows in LDS stash
  const int lnA1 = ((ntl + 1) * 16 + idx);
  const int ntB  = w >> 2;                  // phase B: 1 N-tile per wave
  const int lnB  = LDS_WFZ_ELEMS + (ntB * 16 + idx) * KTOT;

  for (int t = 0; t < SEQ; ++t) {
    const int cur = t & 1, nxt = cur ^ 1;
    const float*  xc_f = x_f32 + cur * (BATCH * HID);
    const __bf16* xc_b = x_bf  + cur * (BATCH * HID);
    float*  xn_f = x_f32 + nxt * (BATCH * HID);
    __bf16* xn_b = x_bf  + nxt * (BATCH * HID);

    // snapshot x_t for the final (parallel) output GEMM
    {
      int g = (wg * 256 + tid) * 8;  // 16*256*8 == BATCH*HID
      *(ulonglong2*)(xhist + (size_t)t * BATCH * HID + g) =
          *(const ulonglong2*)(xc_b + g);
    }

    const int urow = mrow * SEQ + t;
    // prefetch next timestep's u row (the only HBM-streamed operand here)
    __builtin_prefetch(u_bf + (size_t)(urow + 1) * FIN, 0, 1);

    // cache the u_t A-fragments in registers; reused by BOTH phases
    v16bf au[FIN / 32];
#pragma unroll
    for (int kt = 0; kt < FIN / 32; ++kt)
      au[kt] = load_fragA(u_bf, urow, FIN, kt * 32, half);

    // ---- Phase A: fz = sigmoid([u_t, x] @ Wfz + b_fz); 64 cols per WG ----
    const int col0 = wg * 64 + (ntl + 0) * 16 + idx;
    const int col1 = wg * 64 + (ntl + 1) * 16 + idx;
    v8f acc0, acc1;
    {
      float b0 = bias_fz[col0], b1 = bias_fz[col1];
#pragma unroll
      for (int j = 0; j < 8; ++j) { acc0[j] = b0; acc1[j] = b1; }
    }
#pragma unroll
    for (int kt = 0; kt < FIN / 32; ++kt) {          // u section (K = 0..255)
      v16bf b0 = lds_fragB(lnA0 * KTOT + kt * 32, half);
      v16bf b1 = lds_fragB(lnA1 * KTOT + kt * 32, half);
      acc0 = wmma_bf16(au[kt], b0, acc0);
      acc1 = wmma_bf16(au[kt], b1, acc1);
    }
#pragma unroll
    for (int kt = 0; kt < HID / 32; ++kt) {          // x section (K = 256..767)
      v16bf a  = load_fragA(xc_b, mrow, HID, kt * 32, half);
      v16bf b0 = lds_fragB(lnA0 * KTOT + FIN + kt * 32, half);
      v16bf b1 = lds_fragB(lnA1 * KTOT + FIN + kt * 32, half);
      acc0 = wmma_bf16(a, b0, acc0);
      acc1 = wmma_bf16(a, b1, acc1);
    }
#pragma unroll
    for (int j = 0; j < 8; ++j) {
      int m = mt * 16 + j + 8 * half;   // D layout: VGPR j + half -> batch row
      float s0 = 1.0f / (1.0f + expf(-acc0[j]));
      float s1 = 1.0f / (1.0f + expf(-acc1[j]));
      if (col0 < HID) fx_bf[m * HID + col0] = (__bf16)(s0 * xc_f[m * HID + col0]);
      else            z_f32[m * HID + (col0 - HID)] = s0;
      if (col1 < HID) fx_bf[m * HID + col1] = (__bf16)(s1 * xc_f[m * HID + col1]);
      else            z_f32[m * HID + (col1 - HID)] = s1;
    }

    grid_barrier(bar);

    // ---- Phase B: r = tanh([u_t, f*x] @ Wr + b_r); x' = (1-z)x + z r ----
    {
      const int colB = wg * 32 + ntB * 16 + idx;
      v8f acc;
      float bb = bias_r[colB];
#pragma unroll
      for (int j = 0; j < 8; ++j) acc[j] = bb;
#pragma unroll
      for (int kt = 0; kt < FIN / 32; ++kt)          // u section: reuse au[]
        acc = wmma_bf16(au[kt], lds_fragB(lnB + kt * 32, half), acc);
#pragma unroll
      for (int kt = 0; kt < HID / 32; ++kt) {        // f*x section
        v16bf a = load_fragA(fx_bf, mrow, HID, kt * 32, half);
        acc = wmma_bf16(a, lds_fragB(lnB + FIN + kt * 32, half), acc);
      }
#pragma unroll
      for (int j = 0; j < 8; ++j) {
        int m    = mt * 16 + j + 8 * half;
        float r  = tanhf(acc[j]);
        float zz = z_f32[m * HID + colB];
        float xo = xc_f[m * HID + colB];
        float xn = (1.0f - zz) * xo + zz * r;
        xn_f[m * HID + colB] = xn;
        xn_b[m * HID + colB] = (__bf16)xn;
      }
    }

    grid_barrier(bar);
  }
}

// ---------------- final output GEMM: y[b,t,:] = x_t @ w_out + b_out ----------------
__global__ void __launch_bounds__(256) gru_out(
    const __bf16* __restrict__ xhist, const __bf16* __restrict__ WoutT,
    const float* __restrict__ b_out, float* __restrict__ out) {
  const int t    = blockIdx.x;
  const int tid  = threadIdx.x;
  const int w    = tid >> 5;
  const int half = (tid >> 4) & 1;
  const int idx  = tid & 15;
  const int mt   = w & 3;
  const int ntb  = (w >> 2) * 4;    // waves 0-3: N-tiles 0..3, waves 4-7: 4..7
  const __bf16* X = xhist + (size_t)t * BATCH * HID;
  const int mrow  = mt * 16 + idx;
  v8f acc[4];
#pragma unroll
  for (int q = 0; q < 4; ++q) {
    float b = b_out[(ntb + q) * 16 + idx];
#pragma unroll
    for (int j = 0; j < 8; ++j) acc[q][j] = b;
  }
#pragma unroll
  for (int kt = 0; kt < HID / 32; ++kt) {
    v16bf a = load_fragA(X, mrow, HID, kt * 32, half);
#pragma unroll
    for (int q = 0; q < 4; ++q) {
      v16bf b = load_fragB(WoutT, (ntb + q) * 16 + idx, HID, kt * 32, half);
      acc[q] = wmma_bf16(a, b, acc[q]);
    }
  }
#pragma unroll
  for (int q = 0; q < 4; ++q) {
    int n = (ntb + q) * 16 + idx;
#pragma unroll
    for (int j = 0; j < 8; ++j) {
      int m = mt * 16 + j + 8 * half;                  // batch row
      out[((size_t)m * SEQ + t) * FOUT + n] = acc[q][j];
    }
  }
}

// ---------------- host launcher ----------------
extern "C" void kernel_launch(void* const* d_in, const int* in_sizes, int n_in,
                              void* d_out, int out_size, void* d_ws, size_t ws_size,
                              hipStream_t stream) {
  const float* u       = (const float*)d_in[0];
  const float* x0      = (const float*)d_in[1];
  const float* kfz     = (const float*)d_in[2];
  const float* bias_fz = (const float*)d_in[3];
  const float* kr      = (const float*)d_in[4];
  const float* bias_r  = (const float*)d_in[5];
  const float* wout    = (const float*)d_in[6];
  const float* b_out   = (const float*)d_in[7];
  float* out = (float*)d_out;

  // workspace carve-out (all offsets 256B aligned); total ~52 MB
  uint8_t* ws = (uint8_t*)d_ws;
  size_t o = 0;
  auto carve = [&](size_t bytes) -> uint8_t* {
    uint8_t* p = ws + o;
    o = (o + bytes + 255) & ~(size_t)255;
    return p;
  };
  __bf16* u_bf  = (__bf16*)carve((size_t)BATCH * SEQ * FIN * 2);
  __bf16* WfzT  = (__bf16*)carve((size_t)NFZ * KTOT * 2);
  __bf16* WrT   = (__bf16*)carve((size_t)HID * KTOT * 2);
  __bf16* WoutT = (__bf16*)carve((size_t)FOUT * HID * 2);
  __bf16* xhist = (__bf16*)carve((size_t)SEQ * BATCH * HID * 2);
  float*  x_f32 = (float*)carve((size_t)2 * BATCH * HID * 4);
  __bf16* x_bf  = (__bf16*)carve((size_t)2 * BATCH * HID * 2);
  __bf16* fx_bf = (__bf16*)carve((size_t)BATCH * HID * 2);
  float*  z_f32 = (float*)carve((size_t)BATCH * HID * 4);
  int*    bar   = (int*)carve(256);
  (void)ws_size; (void)in_sizes; (void)n_in; (void)out_size;

  // allow 144 KB dynamic LDS (CDNA5: up to 320 KB per workgroup)
  (void)hipFuncSetAttribute((const void*)gru_recurrent,
                            hipFuncAttributeMaxDynamicSharedMemorySize, LDS_BYTES);

  const int nU = BATCH * SEQ * FIN;
  gru_cvt_u<<<(nU / 4 + 255) / 256, 256, 0, stream>>>(u, u_bf, nU);

  const int nW = NFZ * KTOT + HID * KTOT + FOUT * HID;
  gru_prep_w<<<(nW + 255) / 256, 256, 0, stream>>>(kfz, kr, wout, WfzT, WrT, WoutT);

  gru_init_x<<<(BATCH * HID + 255) / 256, 256, 0, stream>>>(x0, x_f32, x_bf, bar);

  gru_recurrent<<<NWG, 256, LDS_BYTES, stream>>>(u_bf, WfzT, WrT, bias_fz, bias_r,
                                                 x_f32, x_bf, fx_bf, z_f32, xhist, bar);

  gru_out<<<SEQ, 256, 0, stream>>>(xhist, WoutT, b_out, out);
}
